// TriangleMultiplicative_89996744721174
// MI455X (gfx1250) — matrix operand order repaired and checked by reading, not compiled, not run on hardware
//
#include <hip/hip_runtime.h>
#include <hip/hip_bf16.h>
#include <stdint.h>

// ---------------------------------------------------------------------------
// CDNA5 (gfx1250) triangle-multiplicative-update, bf16 WMMA pipeline.
// All matrix math uses v_wmma_f32_16x16x32_bf16 (wave32).
// ---------------------------------------------------------------------------

typedef __attribute__((ext_vector_type(16))) __bf16 v16bf;
typedef __attribute__((ext_vector_type(8)))  float  v8f;

union BF16x16 { v16bf v; uint4 q[2]; };

constexpr int  kN   = 384;
constexpr long kNN  = (long)kN * kN;   // 147456 positions
constexpr int  kD   = 128;             // feature dim (== C)
constexpr int  kLDZ = 136;             // padded LDS row stride in halves

__device__ __forceinline__ unsigned short f2bf(float f) {
  unsigned int u = __float_as_uint(f);
  u += 0x7FFFu + ((u >> 16) & 1u);     // round-to-nearest-even
  return (unsigned short)(u >> 16);
}
__device__ __forceinline__ float bf2f(unsigned short h) {
  return __uint_as_float((unsigned int)h << 16);
}
__device__ __forceinline__ float sigmoidf_(float x) { return 1.0f / (1.0f + __expf(-x)); }

// A operand (16x32 bf16). lanes 0-15: M=lane, halves K={0..7},{16..23};
// lanes 16-31: M=lane-16, halves K={8..15},{24..31}. Two 16B chunks/lane.
__device__ __forceinline__ v16bf load_a16x32(const unsigned short* base, int ld, int lane) {
  const int m  = lane & 15;
  const int ks = (lane & 16) ? 8 : 0;
  const unsigned short* p = base + (long)m * ld + ks;
  BF16x16 u;
  u.q[0] = *(const uint4*)(p);
  u.q[1] = *(const uint4*)(p + 16);
  return u.v;
}

// B operand (32x16 bf16) for D = A * Mᵀ with M stored row-major [n][k]:
// lane n holds K={0..7},{16..23} (lanes 0-15) or {8..15},{24..31} (lanes 16-31).
__device__ __forceinline__ v16bf load_bT32x16(const unsigned short* base, int ld, int lane) {
  const int n  = lane & 15;
  const int kb = (lane & 16) ? 8 : 0;
  const unsigned short* p = base + (long)n * ld + kb;
  BF16x16 u;
  u.q[0] = *(const uint4*)(p);
  u.q[1] = *(const uint4*)(p + 16);
  return u.v;
}

__device__ __forceinline__ v8f wmma_bf16(v16bf a, v16bf b, v8f c) {
  return __builtin_amdgcn_wmma_f32_16x16x32_bf16(false, a, false, b, (short)0, c, false, false);
}

// Pack 8 floats (consecutive positions) into a uint4 of bf16 for one b128 store.
__device__ __forceinline__ uint4 pack8bf(const float* v) {
  unsigned int w[4];
#pragma unroll
  for (int j = 0; j < 4; ++j)
    w[j] = (unsigned int)f2bf(v[2 * j]) | ((unsigned int)f2bf(v[2 * j + 1]) << 16);
  return make_uint4(w[0], w[1], w[2], w[3]);
}

// ---------------------------------------------------------------------------
// Kernel 0: convert the six [128x128] torch-layout weight matrices to bf16.
// Order in Wc: [Wag, Wa, Wbg, Wb, Wgg, Wg]
// ---------------------------------------------------------------------------
__global__ void tri_prep_weights(const float* __restrict__ wag, const float* __restrict__ wa,
                                 const float* __restrict__ wbg, const float* __restrict__ wb,
                                 const float* __restrict__ wgg, const float* __restrict__ wg,
                                 unsigned short* __restrict__ Wc) {
  int idx = blockIdx.x * blockDim.x + threadIdx.x;
  if (idx >= 6 * kD * kD) return;
  int m = idx >> 14;
  int r = idx & 16383;
  const float* src[6] = {wag, wa, wbg, wb, wgg, wg};
  Wc[idx] = f2bf(src[m][r]);
}

// ---------------------------------------------------------------------------
// Kernel 1: fused LayerNorm + 5 projections + sigmoid gating.
// Block = 128 threads (4 waves) handles 16 positions.
// Outputs (ALL channel-major [ch][pos] bf16, written as packed b128):
//   a_t, b_t  : gated projections for the triangle GEMM
//   gbuf      : sigmoid output gate
// ---------------------------------------------------------------------------
__global__ void __launch_bounds__(128)
tri_proj(const float* __restrict__ zij, const float* __restrict__ gam, const float* __restrict__ bet,
         const unsigned short* __restrict__ Wc,
         const float* __restrict__ bag, const float* __restrict__ ba,
         const float* __restrict__ bbg, const float* __restrict__ bb,
         const float* __restrict__ bgg,
         unsigned short* __restrict__ a_t, unsigned short* __restrict__ b_t,
         unsigned short* __restrict__ gbuf) {
  __shared__ unsigned short zt[16 * kLDZ];
  const int t = threadIdx.x;
  const int lane = t & 31;
  const int wv = __builtin_amdgcn_readfirstlane(t >> 5);  // scalar wave id
  const long pos0 = (long)blockIdx.x * 16;

  // ---- LayerNorm of 16 rows x 128, 8 lanes per row ----
  {
    const int r = t >> 3;
    const int s = t & 7;
    const float* row = zij + (pos0 + r) * kD + s * 16;
    float x[16];
    float4 t0 = ((const float4*)row)[0];
    float4 t1 = ((const float4*)row)[1];
    float4 t2 = ((const float4*)row)[2];
    float4 t3 = ((const float4*)row)[3];
    x[0]=t0.x; x[1]=t0.y; x[2]=t0.z; x[3]=t0.w;
    x[4]=t1.x; x[5]=t1.y; x[6]=t1.z; x[7]=t1.w;
    x[8]=t2.x; x[9]=t2.y; x[10]=t2.z; x[11]=t2.w;
    x[12]=t3.x; x[13]=t3.y; x[14]=t3.z; x[15]=t3.w;
    float sum = 0.f, sq = 0.f;
#pragma unroll
    for (int j = 0; j < 16; ++j) { sum += x[j]; sq += x[j] * x[j]; }
#pragma unroll
    for (int msk = 1; msk < 8; msk <<= 1) {
      sum += __shfl_xor(sum, msk, 8);
      sq  += __shfl_xor(sq,  msk, 8);
    }
    const float mean = sum * (1.0f / kD);
    const float var  = sq * (1.0f / kD) - mean * mean;
    const float rstd = __frsqrt_rn(var + 1e-5f);
    unsigned int w[8];
#pragma unroll
    for (int j = 0; j < 8; ++j) {
      const int c0 = s * 16 + 2 * j;
      const unsigned short h0 = f2bf((x[2*j]   - mean) * rstd * gam[c0]   + bet[c0]);
      const unsigned short h1 = f2bf((x[2*j+1] - mean) * rstd * gam[c0+1] + bet[c0+1]);
      w[j] = (unsigned int)h0 | ((unsigned int)h1 << 16);
    }
    uint4* dst = (uint4*)&zt[r * kLDZ + s * 16];
    dst[0] = make_uint4(w[0], w[1], w[2], w[3]);
    dst[1] = make_uint4(w[4], w[5], w[6], w[7]);
  }
  __syncthreads();

  // Preload A operand (16 positions x 128 K) once; reuse for all jobs.
  v16bf Az[4];
#pragma unroll
  for (int ks = 0; ks < 4; ++ks) Az[ks] = load_a16x32(&zt[ks * 32], kLDZ, lane);

  const int mb = (lane & 16) ? 8 : 0;
  // 24 fused jobs (scalar control): kind 0 = a, 1 = b, 2 = g; 8 col-chunks each.
  for (int job = wv; job < 24; job += 4) {
    const int kind  = job >> 3;
    const int chunk = job & 7;
    const unsigned short* wgate = Wc + (long)(kind * 2) * 16384 + chunk * 16 * kD;
    v8f accg = (v8f)0.0f;
    v8f accv = (v8f)0.0f;
    if (kind < 2) {
      const unsigned short* wval = wgate + 16384;
      // Batch ALL 8 B-tile loads before any WMMA: one long load clause whose
      // latency overlaps WMMA issue via partial loadcnt waits.
      v16bf Bg[4], Bv[4];
#pragma unroll
      for (int ks = 0; ks < 4; ++ks) Bg[ks] = load_bT32x16(wgate + ks * 32, kD, lane);
#pragma unroll
      for (int ks = 0; ks < 4; ++ks) Bv[ks] = load_bT32x16(wval + ks * 32, kD, lane);
#pragma unroll
      for (int ks = 0; ks < 4; ++ks) {
        accg = wmma_bf16(Az[ks], Bg[ks], accg);
        accv = wmma_bf16(Az[ks], Bv[ks], accv);
      }
    } else {
      v16bf Bg[4];
#pragma unroll
      for (int ks = 0; ks < 4; ++ks) Bg[ks] = load_bT32x16(wgate + ks * 32, kD, lane);
#pragma unroll
      for (int ks = 0; ks < 4; ++ks) accg = wmma_bf16(Az[ks], Bg[ks], accg);
    }
    const int o = chunk * 16 + (lane & 15);
    if (kind < 2) {
      const float bv = (kind == 0) ? ba[o]  : bb[o];
      const float bg = (kind == 0) ? bag[o] : bbg[o];
      unsigned short* dst = (kind == 0) ? a_t : b_t;
      float v[8];
#pragma unroll
      for (int r = 0; r < 8; ++r)
        v[r] = (accv[r] + bv) * sigmoidf_(accg[r] + bg);
      *(uint4*)(dst + (long)o * kNN + pos0 + mb) = pack8bf(v);   // one b128 store
    } else {
      const float bg = bgg[o];
      float v[8];
#pragma unroll
      for (int r = 0; r < 8; ++r)
        v[r] = sigmoidf_(accg[r] + bg);
      *(uint4*)(gbuf + (long)o * kNN + pos0 + mb) = pack8bf(v);
    }
  }
}

// ---------------------------------------------------------------------------
// Kernel 2: triangle einsum as 128-channel batched GEMM: ab_ch = A_ch · B_chᵀ.
// Block = 256 threads (8 waves, 4x2): 128(i) x 128(e) tile; each wave a 32x64
// register tile (2x4 accumulators, 8 WMMAs / K-step vs 6 operand loads).
// K-loop is double-buffered so loads overlap WMMA issue. Output stored
// channel-major [ch][i][e] -> coalesced across lanes (contiguous along e).
// ---------------------------------------------------------------------------
__global__ void __launch_bounds__(256)
tri_gemm(const unsigned short* __restrict__ a_t, const unsigned short* __restrict__ b_t,
         unsigned short* __restrict__ abuf) {
  const int t = threadIdx.x;
  const int lane = t & 31;
  const int wave = __builtin_amdgcn_readfirstlane(t >> 5);
  const int wi = wave >> 1;          // 0..3
  const int we = wave & 1;           // 0..1
  const int ch = blockIdx.z;
  const int ibase = blockIdx.x * 128 + wi * 32;
  const int ebase = blockIdx.y * 128 + we * 64;
  const unsigned short* A = a_t + (long)ch * kNN + (long)ibase * kN;
  const unsigned short* B = b_t + (long)ch * kNN + (long)ebase * kN;

  v8f acc[2][4];
#pragma unroll
  for (int p = 0; p < 2; ++p)
#pragma unroll
    for (int q = 0; q < 4; ++q) acc[p][q] = (v8f)0.0f;

  v16bf cA[2], cB[4];
#pragma unroll
  for (int p = 0; p < 2; ++p) cA[p] = load_a16x32(A + (long)(p * 16) * kN, kN, lane);
#pragma unroll
  for (int q = 0; q < 4; ++q) cB[q] = load_bT32x16(B + (long)(q * 16) * kN, kN, lane);

#pragma unroll 2
  for (int k0 = 32; k0 < kN; k0 += 32) {
    v16bf nA[2], nB[4];
#pragma unroll
    for (int p = 0; p < 2; ++p) nA[p] = load_a16x32(A + (long)(p * 16) * kN + k0, kN, lane);
#pragma unroll
    for (int q = 0; q < 4; ++q) nB[q] = load_bT32x16(B + (long)(q * 16) * kN + k0, kN, lane);
#pragma unroll
    for (int p = 0; p < 2; ++p)
#pragma unroll
      for (int q = 0; q < 4; ++q) acc[p][q] = wmma_bf16(cA[p], cB[q], acc[p][q]);
#pragma unroll
    for (int p = 0; p < 2; ++p) cA[p] = nA[p];
#pragma unroll
    for (int q = 0; q < 4; ++q) cB[q] = nB[q];
  }
#pragma unroll
  for (int p = 0; p < 2; ++p)
#pragma unroll
    for (int q = 0; q < 4; ++q) acc[p][q] = wmma_bf16(cA[p], cB[q], acc[p][q]);

  // Store channel-major: lanes 0-15 / 16-31 each cover 32B contiguous in e.
  const int n  = lane & 15;
  const int mb = (lane & 16) ? 8 : 0;
  unsigned short* out = abuf + (long)ch * kNN;
#pragma unroll
  for (int p = 0; p < 2; ++p)
#pragma unroll
    for (int q = 0; q < 4; ++q) {
#pragma unroll
      for (int r = 0; r < 8; ++r) {
        const long i = ibase + p * 16 + mb + r;
        const long e = ebase + q * 16 + n;
        out[i * kN + e] = f2bf(acc[p][q][r]);
      }
    }
}

// ---------------------------------------------------------------------------
// Kernel 3: fused LayerNorm(ab) + output projection (Wg) + gate multiply.
// Block = 128 threads handles 16 consecutive positions (same i row, e-range).
// ab and g are channel-major; transpose ab through LDS (each thread loads one
// contiguous 32B chunk per channel), LN in place, then WMMA.
// ---------------------------------------------------------------------------
__global__ void __launch_bounds__(128)
tri_out(const unsigned short* __restrict__ abuf,
        const float* __restrict__ gam, const float* __restrict__ bet,
        const unsigned short* __restrict__ Wc,     // Wg at slot 5
        const float* __restrict__ wgb,
        const unsigned short* __restrict__ gbuf,
        float* __restrict__ out) {
  __shared__ unsigned short zt[16 * kLDZ];
  const int t = threadIdx.x;
  const int lane = t & 31;
  const int wv = __builtin_amdgcn_readfirstlane(t >> 5);
  const long pos0 = (long)blockIdx.x * 16;

  // ---- phase 0: gather channel-major ab into LDS transposed [pos][ch] ----
  {
    const int ch = t;                       // 128 threads == 128 channels
    const unsigned short* p = abuf + (long)ch * kNN + pos0;
    uint4 q0 = ((const uint4*)p)[0];
    uint4 q1 = ((const uint4*)p)[1];
    unsigned int w[8] = {q0.x, q0.y, q0.z, q0.w, q1.x, q1.y, q1.z, q1.w};
#pragma unroll
    for (int j = 0; j < 8; ++j) {
      zt[(2 * j)     * kLDZ + ch] = (unsigned short)(w[j] & 0xFFFFu);
      zt[(2 * j + 1) * kLDZ + ch] = (unsigned short)(w[j] >> 16);
    }
  }
  __syncthreads();

  // ---- phase 1: LayerNorm rows in place (each thread owns its 16 elems) ----
  {
    const int r = t >> 3;
    const int s = t & 7;
    unsigned short* row = &zt[r * kLDZ + s * 16];
    uint4 q0 = ((uint4*)row)[0];
    uint4 q1 = ((uint4*)row)[1];
    unsigned int w[8] = {q0.x, q0.y, q0.z, q0.w, q1.x, q1.y, q1.z, q1.w};
    float x[16];
#pragma unroll
    for (int j = 0; j < 8; ++j) {
      x[2*j]   = bf2f((unsigned short)(w[j] & 0xFFFFu));
      x[2*j+1] = bf2f((unsigned short)(w[j] >> 16));
    }
    float sum = 0.f, sq = 0.f;
#pragma unroll
    for (int j = 0; j < 16; ++j) { sum += x[j]; sq += x[j] * x[j]; }
#pragma unroll
    for (int msk = 1; msk < 8; msk <<= 1) {
      sum += __shfl_xor(sum, msk, 8);
      sq  += __shfl_xor(sq,  msk, 8);
    }
    const float mean = sum * (1.0f / kD);
    const float var  = sq * (1.0f / kD) - mean * mean;
    const float rstd = __frsqrt_rn(var + 1e-5f);
    unsigned int ww[8];
#pragma unroll
    for (int j = 0; j < 8; ++j) {
      const int c0 = s * 16 + 2 * j;
      const unsigned short h0 = f2bf((x[2*j]   - mean) * rstd * gam[c0]   + bet[c0]);
      const unsigned short h1 = f2bf((x[2*j+1] - mean) * rstd * gam[c0+1] + bet[c0+1]);
      ww[j] = (unsigned int)h0 | ((unsigned int)h1 << 16);
    }
    ((uint4*)row)[0] = make_uint4(ww[0], ww[1], ww[2], ww[3]);
    ((uint4*)row)[1] = make_uint4(ww[4], ww[5], ww[6], ww[7]);
  }
  __syncthreads();

  v16bf Az[4];
#pragma unroll
  for (int ks = 0; ks < 4; ++ks) Az[ks] = load_a16x32(&zt[ks * 32], kLDZ, lane);

  const unsigned short* W5 = Wc + 5L * 16384;
  const int mb = (lane & 16) ? 8 : 0;
  for (int chunk = wv; chunk < 8; chunk += 4) {
    // Batch the 4 weight-tile loads before the WMMA chain.
    v16bf Bw[4];
#pragma unroll
    for (int ks = 0; ks < 4; ++ks)
      Bw[ks] = load_bT32x16(W5 + chunk * 16 * kD + ks * 32, kD, lane);
    v8f acc = (v8f)0.0f;
#pragma unroll
    for (int ks = 0; ks < 4; ++ks) acc = wmma_bf16(Az[ks], Bw[ks], acc);
    const int o = chunk * 16 + (lane & 15);
    const float bias = wgb[o];
    // channel-major gate: one contiguous b128 load per lane
    uint4 gq = *(const uint4*)(gbuf + (long)o * kNN + pos0 + mb);
    unsigned int gw[4] = {gq.x, gq.y, gq.z, gq.w};
#pragma unroll
    for (int r = 0; r < 8; ++r) {
      const unsigned short gh = (r & 1) ? (unsigned short)(gw[r >> 1] >> 16)
                                        : (unsigned short)(gw[r >> 1] & 0xFFFFu);
      const long pos = pos0 + mb + r;
      out[pos * kD + o] = bf2f(gh) * (acc[r] + bias);
    }
  }
}

// ---------------------------------------------------------------------------
// Launch. Workspace layout (bytes):
//   [0, 256KB)  : 6 bf16 weight matrices
//   + 37.75MB   : a_t  [128][147456] bf16 (channel-major)
//   + 37.75MB   : b_t  [128][147456] bf16
//   + 37.75MB   : gbuf [128][147456] bf16 (channel-major)
//   + 37.75MB   : abuf [128][147456] bf16 (channel-major)
// Total ≈ 151.3 MB (intermediates stay L2-resident: 192MB L2).
// ---------------------------------------------------------------------------
extern "C" void kernel_launch(void* const* d_in, const int* in_sizes, int n_in,
                              void* d_out, int out_size, void* d_ws, size_t ws_size,
                              hipStream_t stream) {
  (void)in_sizes; (void)n_in; (void)out_size; (void)ws_size;
  const float* zij   = (const float*)d_in[0];
  const float* n1g   = (const float*)d_in[1];
  const float* n1b   = (const float*)d_in[2];
  const float* Wag_w = (const float*)d_in[3];
  const float* Wag_b = (const float*)d_in[4];
  const float* Wa_w  = (const float*)d_in[5];
  const float* Wa_b  = (const float*)d_in[6];
  const float* Wbg_w = (const float*)d_in[7];
  const float* Wbg_b = (const float*)d_in[8];
  const float* Wb_w  = (const float*)d_in[9];
  const float* Wb_b  = (const float*)d_in[10];
  const float* n2g   = (const float*)d_in[11];
  const float* n2b   = (const float*)d_in[12];
  const float* Wgg_w = (const float*)d_in[13];
  const float* Wgg_b = (const float*)d_in[14];
  const float* Wg_w  = (const float*)d_in[15];
  const float* Wg_b  = (const float*)d_in[16];

  char* ws = (char*)d_ws;
  unsigned short* Wc   = (unsigned short*)ws;
  unsigned short* a_t  = (unsigned short*)(ws + (1 << 18));
  unsigned short* b_t  = a_t + 128L * kNN;
  unsigned short* gbuf = b_t + 128L * kNN;
  unsigned short* abuf = gbuf + 128L * kNN;

  tri_prep_weights<<<384, 256, 0, stream>>>(Wag_w, Wa_w, Wbg_w, Wb_w, Wgg_w, Wg_w, Wc);
  tri_proj<<<(int)(kNN / 16), 128, 0, stream>>>(zij, n1g, n1b, Wc,
                                                Wag_b, Wa_b, Wbg_b, Wb_b, Wgg_b,
                                                a_t, b_t, gbuf);
  tri_gemm<<<dim3(kN / 128, kN / 128, kD), 256, 0, stream>>>(a_t, b_t, abuf);
  tri_out<<<(int)(kNN / 16), 128, 0, stream>>>(abuf, n2g, n2b, Wc, Wg_b, gbuf,
                                               (float*)d_out);
}
